// Discriminator_20177756357013
// MI455X (gfx1250) — matrix operand order repaired and checked by reading, not compiled, not run on hardware
//
#include <hip/hip_runtime.h>
#include <hip/hip_bf16.h>
#include <math.h>

// ---------------------------------------------------------------------------
// Types for CDNA5 WMMA (wave32). 16x16x32 bf16 -> f32 accumulate.
// ---------------------------------------------------------------------------
typedef __attribute__((ext_vector_type(16))) __bf16 v16bf;
typedef __attribute__((ext_vector_type(8)))  __bf16 v8bf;
typedef __attribute__((ext_vector_type(8)))  float  v8f;
typedef __attribute__((ext_vector_type(4)))  unsigned int u32x4;
typedef __attribute__((ext_vector_type(8)))  unsigned int u32x8;

#define BB 16384
#define EE 512
#define HH 1024
#define GG (4 * HH)
#define TT 64
#define KC 256            // K-chunk staged in LDS
#define SL (KC + 8)       // LDS row stride in elements (+16B pad: bank-conflict free)

__device__ __forceinline__ float sigf(float x) { return 1.0f / (1.0f + __expf(-x)); }

__device__ __forceinline__ v8f wmma_bf16(v16bf a, v16bf b, v8f c) {
    return __builtin_amdgcn_wmma_f32_16x16x32_bf16(false, a, false, b, (short)0, c,
                                                   false, false);
}

// A fragment: 16x32 bf16 tile of row-major A, origin (row0,k0). ISA 7.12.2 layout:
// lane L: r=L&15, hi=L>>4; VGPR0..3 = K hi*8+0..7, VGPR4..7 = K 16+hi*8+0..7.
__device__ __forceinline__ v16bf load_frag_a(const __bf16* __restrict__ A, int lda,
                                             int row0, int k0) {
    int lane = threadIdx.x & 31;
    int r = lane & 15, hi = lane >> 4;
    const __bf16* p = A + (size_t)(row0 + r) * lda + k0 + hi * 8;
    v8bf lo = *(const v8bf*)(p);
    v8bf up = *(const v8bf*)(p + 16);
    v16bf f;
#pragma unroll
    for (int t = 0; t < 8; ++t) { f[t] = lo[t]; f[t + 8] = up[t]; }
    return f;
}

// B fragment read out of an LDS-staged N-panel (row stride SL elements).
// Lane L: column n=L&15, K (L>>4)*8+0..7 and 16+(L>>4)*8+0..7 -> 2x ds_load_b128.
__device__ __forceinline__ v16bf load_frag_b_lds(const __bf16* s, int n0, int k0) {
    int lane = threadIdx.x & 31;
    int n = lane & 15, kst = (lane >> 4) * 8;
    const __bf16* p = s + (n0 + n) * SL + k0 + kst;
    v8bf lo = *(const v8bf*)(p);
    v8bf up = *(const v8bf*)(p + 16);
    v16bf f;
#pragma unroll
    for (int t = 0; t < 8; ++t) { f[t] = lo[t]; f[t + 8] = up[t]; }
    return f;
}

// ---------------------------------------------------------------------------
// Tensor Data Mover: load a (rows x KC) bf16 tile (row stride EE elements in
// global) into LDS at lds_off, padding each 512B row by 16B so LDS row stride
// is SL elements. D# layout per CDNA5 ISA.
// ---------------------------------------------------------------------------
__device__ __forceinline__ void tdm_load_tile(unsigned lds_off,
                                              const __bf16* gsrc, int rows) {
    unsigned long long ga = (unsigned long long)(const void*)gsrc;
    u32x4 g0;
    g0[0] = 1u;                                        // count=1, user descriptor
    g0[1] = lds_off;                                   // lds_addr (bytes)
    g0[2] = (unsigned)ga;                              // global_addr[31:0]
    g0[3] = ((unsigned)(ga >> 32) & 0x01FFFFFFu)       // global_addr[56:32]
          | 0x80000000u;                               // type = 2 ("image")
    u32x8 g1;
    g1[0] = (1u << 16)                                 // data_size = 2 bytes
          | (1u << 20)                                 // pad_enable
          | (6u << 22)                                 // pad_interval: 64x8B = 512B row
          | (3u << 25);                                // pad_amount: 4 DWORDs = 16B
    g1[1] = ((unsigned)KC) << 16;                      // tensor_dim0[15:0] = KC
    g1[2] = 0x4000u << 16;                             // tensor_dim1 = 16384 rows
    g1[3] = ((unsigned)KC) << 16;                      // tile_dim0 = KC
    g1[4] = (unsigned)rows;                            // tile_dim1 (tile_dim2 = 0)
    g1[5] = (unsigned)EE;                              // tensor_dim0_stride (elems)
    g1[6] = 0u;
    g1[7] = 0u;
    asm volatile("tensor_load_to_lds %0, %1" :: "s"(g0), "s"(g1) : "memory");
}

// ---------------------------------------------------------------------------
// Generic bf16 WMMA GEMM: C[M x N] = A[M x K] @ W[N x K]^T + bias.
// Block = 8 waves (2M x 4N), block tile 32 x 64. B panel staged in LDS per
// 256-wide K chunk (global_load_b128 -> ds_store_b128), WMMA feeds from LDS.
// grid.x = N/64, grid.y = M/32.
// ---------------------------------------------------------------------------
__global__ __launch_bounds__(256) void gemm_bf16_wmma(
    const __bf16* __restrict__ A, int lda,
    const __bf16* __restrict__ W, int ldw,
    __bf16* __restrict__ C, int ldc,
    const float* __restrict__ bias, int K) {
    __shared__ __bf16 sB[64 * SL];                     // 33 KB
    int wave = threadIdx.x >> 5;
    int lane = threadIdx.x & 31;
    int row0  = blockIdx.y * 32 + (wave >> 2) * 16;
    int col0w = (wave & 3) * 16;
    int n0    = blockIdx.x * 64;

    v8f acc = {};
    for (int c = 0; c < K; c += KC) {
        if (c) __syncthreads();                        // protect LDS reuse
        for (int idx = threadIdx.x; idx < 64 * (KC / 8); idx += 256) {
            int r = idx / (KC / 8), q = idx % (KC / 8);
            *(v8bf*)(sB + r * SL + q * 8) =
                *(const v8bf*)(W + (size_t)(n0 + r) * ldw + c + q * 8);
        }
        __syncthreads();
#pragma unroll
        for (int k0 = 0; k0 < KC; k0 += 32) {
            __builtin_prefetch(A + (size_t)(row0 + (lane & 15)) * lda + c + k0 + 32,
                               0, 1);
            v16bf a = load_frag_a(A, lda, row0, c + k0);
            v16bf b = load_frag_b_lds(sB, col0w, k0);
            acc = wmma_bf16(a, b, acc);
        }
    }

    int n = lane & 15, mh = (lane >> 4) * 8;
    float bv = bias ? bias[n0 + col0w + n] : 0.0f;
#pragma unroll
    for (int r = 0; r < 8; ++r)
        C[(size_t)(row0 + mh + r) * ldc + n0 + col0w + n] = (__bf16)(acc[r] + bv);
}

// ---------------------------------------------------------------------------
// Fused LSTM step: gates = x @ W_ih^T with 4 gate accumulators per wave.
// The four 16-row W_ih gate tiles are DMA'd into LDS by the Tensor Data Mover
// (wave 0 issues, s_wait_tensorcnt, barrier publishes). Epilogue adds the
// loop-invariant gates_h (bf16, L2-resident), applies the cell nonlinearity
// against loop-invariant c0, writes h (bf16).
// Block = 8 waves, each a 16-row stripe on one 16-col tile j0 of h.
// grid.x = H/16, grid.y = B/128.
// ---------------------------------------------------------------------------
__global__ __launch_bounds__(256) void lstm_step(
    const __bf16* __restrict__ x,    // B x E
    const __bf16* __restrict__ Wih,  // 4H x E (row-major N x K)
    const __bf16* __restrict__ gh,   // B x 4H
    const __bf16* __restrict__ c0,   // B x H
    __bf16* __restrict__ h) {        // B x H out
    __shared__ __bf16 sW[64 * SL];                     // 4 gates x 16 rows, 33 KB
    int wave = threadIdx.x >> 5;
    int lane = threadIdx.x & 31;
    int row0 = blockIdx.y * 128 + wave * 16;
    int j0   = blockIdx.x * 16;

    v8f ai = {}, af = {}, ag = {}, ao = {};
    for (int c = 0; c < EE / KC; ++c) {
        if (c) __syncthreads();
        if (threadIdx.x < 32) {                        // wave 0 drives the TDM
            unsigned lbase = (unsigned)(size_t)(void*)sW;
#pragma unroll
            for (int g = 0; g < 4; ++g)
                tdm_load_tile(lbase + (unsigned)(g * 16 * SL * 2),
                              Wih + (size_t)(g * HH + j0) * EE + c * KC, 16);
            __builtin_amdgcn_s_wait_tensorcnt(0);
        }
        __syncthreads();
#pragma unroll
        for (int k0 = 0; k0 < KC; k0 += 32) {
            v16bf a  = load_frag_a(x, EE, row0, c * KC + k0);
            v16bf bi = load_frag_b_lds(sW, 0,  k0);
            v16bf bf = load_frag_b_lds(sW, 16, k0);
            v16bf bg = load_frag_b_lds(sW, 32, k0);
            v16bf bo = load_frag_b_lds(sW, 48, k0);
            ai = wmma_bf16(a, bi, ai);
            af = wmma_bf16(a, bf, af);
            ag = wmma_bf16(a, bg, ag);
            ao = wmma_bf16(a, bo, ao);
        }
    }

    int n  = lane & 15;
    int mh = (lane >> 4) * 8;
#pragma unroll
    for (int r = 0; r < 8; ++r) {
        int row = row0 + mh + r;
        const __bf16* g = gh + (size_t)row * GG + j0 + n;
        float gi = ai[r] + (float)g[0];
        float gf = af[r] + (float)g[HH];
        float gg = ag[r] + (float)g[2 * HH];
        float go = ao[r] + (float)g[3 * HH];
        float cc = sigf(gf) * (float)c0[(size_t)row * HH + j0 + n] +
                   sigf(gi) * tanhf(gg);
        float hv = sigf(go) * tanhf(cc);
        h[(size_t)row * HH + j0 + n] = (__bf16)hv;
    }
}

// ---------------------------------------------------------------------------
// Final head: logits = h @ final_W^T + final_b (N=2), softmax, fp32 out.
// ---------------------------------------------------------------------------
__global__ __launch_bounds__(256) void final_softmax(
    const __bf16* __restrict__ h, const float* __restrict__ Wf,
    const float* __restrict__ bf, float* __restrict__ out) {
    int wave = threadIdx.x >> 5;
    int lane = threadIdx.x & 31;
    int row = blockIdx.x * 8 + wave;
    const __bf16* hr = h + (size_t)row * HH;
    float s0 = 0.0f, s1 = 0.0f;
    for (int k = lane; k < HH; k += 32) {
        float hv = (float)hr[k];
        s0 += hv * Wf[k];
        s1 += hv * Wf[HH + k];
    }
#pragma unroll
    for (int off = 16; off > 0; off >>= 1) {
        s0 += __shfl_xor(s0, off, 32);
        s1 += __shfl_xor(s1, off, 32);
    }
    if (lane == 0) {
        float l0 = s0 + bf[0], l1 = s1 + bf[1];
        float mx = fmaxf(l0, l1);
        float e0 = __expf(l0 - mx), e1 = __expf(l1 - mx);
        float inv = 1.0f / (e0 + e1);
        out[(size_t)row * 2 + 0] = e0 * inv;
        out[(size_t)row * 2 + 1] = e1 * inv;
    }
}

// ---------------------------------------------------------------------------
// Helpers: fp32 -> bf16 conversion; bias combine
// ---------------------------------------------------------------------------
__global__ void cvt_f32_bf16(const float* __restrict__ in, __bf16* __restrict__ out,
                             int n) {
    int i = blockIdx.x * blockDim.x + threadIdx.x;
    if (i < n) out[i] = (__bf16)in[i];
}

__global__ void bias_combine(const float* __restrict__ a, const float* __restrict__ b,
                             float* __restrict__ o, int n) {
    int i = blockIdx.x * blockDim.x + threadIdx.x;
    if (i < n) o[i] = a[i] + b[i];
}

// ---------------------------------------------------------------------------
// Host launcher
// ---------------------------------------------------------------------------
extern "C" void kernel_launch(void* const* d_in, const int* in_sizes, int n_in,
                              void* d_out, int out_size, void* d_ws, size_t ws_size,
                              hipStream_t stream) {
    const float* start_emb = (const float*)d_in[0];
    const float* h0        = (const float*)d_in[1];
    const float* c0        = (const float*)d_in[2];
    const float* W_ih      = (const float*)d_in[3];
    const float* W_hh      = (const float*)d_in[4];
    const float* b_ih      = (const float*)d_in[5];
    const float* b_hh      = (const float*)d_in[6];
    const float* fc_W      = (const float*)d_in[7];
    const float* fc_b      = (const float*)d_in[8];
    const float* final_W   = (const float*)d_in[9];
    const float* final_b   = (const float*)d_in[10];
    float* out = (float*)d_out;

    char* ws = (char*)d_ws;
    size_t off = 0;
    auto alloc = [&](size_t bytes) {
        void* p = ws + off;
        off = (off + bytes + 255) & ~(size_t)255;
        return p;
    };
    __bf16* gh    = (__bf16*)alloc((size_t)BB * GG * 2);   // 128 MB, L2-resident
    __bf16* xb    = (__bf16*)alloc((size_t)BB * EE * 2);   // 16 MB
    __bf16* hb    = (__bf16*)alloc((size_t)BB * HH * 2);   // 32 MB (h0b then h)
    __bf16* c0b   = (__bf16*)alloc((size_t)BB * HH * 2);   // 32 MB
    __bf16* wihb  = (__bf16*)alloc((size_t)GG * EE * 2);   // 4 MB
    __bf16* whhb  = (__bf16*)alloc((size_t)GG * HH * 2);   // 8 MB
    __bf16* fcwb  = (__bf16*)alloc((size_t)EE * HH * 2);   // 1 MB
    float*  biasc = (float*)alloc((size_t)GG * 4);
    (void)ws_size; (void)in_sizes; (void)n_in; (void)out_size;

    auto cgrid = [](int n) { return dim3((n + 255) / 256); };

    // 1) Precision conversions (one-time)
    cvt_f32_bf16<<<cgrid(BB * EE), 256, 0, stream>>>(start_emb, xb, BB * EE);
    cvt_f32_bf16<<<cgrid(BB * HH), 256, 0, stream>>>(h0, hb, BB * HH);
    cvt_f32_bf16<<<cgrid(BB * HH), 256, 0, stream>>>(c0, c0b, BB * HH);
    cvt_f32_bf16<<<cgrid(GG * EE), 256, 0, stream>>>(W_ih, wihb, GG * EE);
    cvt_f32_bf16<<<cgrid(GG * HH), 256, 0, stream>>>(W_hh, whhb, GG * HH);
    cvt_f32_bf16<<<cgrid(EE * HH), 256, 0, stream>>>(fc_W, fcwb, EE * HH);
    bias_combine<<<cgrid(GG), 256, 0, stream>>>(b_ih, b_hh, biasc, GG);

    // 2) Loop-invariant gates_h = h0 @ W_hh^T + (b_ih + b_hh)   [M=B, N=4H, K=H]
    gemm_bf16_wmma<<<dim3(GG / 64, BB / 32), 256, 0, stream>>>(
        hb, HH, whhb, HH, gh, GG, biasc, HH);

    // 3) 64 fused LSTM steps
    for (int t = 0; t < TT; ++t) {
        lstm_step<<<dim3(HH / 16, BB / 128), 256, 0, stream>>>(xb, wihb, gh, c0b, hb);
        if (t + 1 < TT) {  // x_next = h @ fc_W^T + fc_b   [M=B, N=E, K=H]
            gemm_bf16_wmma<<<dim3(EE / 64, BB / 32), 256, 0, stream>>>(
                hb, HH, fcwb, HH, xb, EE, fc_b, HH);
        }
    }

    // 4) logits + softmax
    final_softmax<<<dim3(BB / 8), 256, 0, stream>>>(hb, final_W, final_b, out);
}
